// SimpleBaseline_87393994539814
// MI455X (gfx1250) — matrix operand order repaired and checked by reading, not compiled, not run on hardware
//
#include <hip/hip_runtime.h>
#include <math.h>

typedef __bf16 bf16;
typedef __bf16 v16bf __attribute__((ext_vector_type(16)));
typedef __bf16 v8bf  __attribute__((ext_vector_type(8)));
typedef float  v8f   __attribute__((ext_vector_type(8)));

#define ACT_NONE 0
#define ACT_RELU 1
#define ACT_TANH 2

typedef __attribute__((address_space(3))) bf16 lds_bf16;

// Async copy 16B/lane global -> LDS (CDNA5, tracked by ASYNCcnt).
__device__ __forceinline__ void async_copy_b128(lds_bf16* lds, const bf16* g)
{
#if defined(__gfx1250__)
    asm volatile("global_load_async_to_lds_b128 %0, %1, off"
                 :: "v"(lds), "v"(g) : "memory");
#endif
}

template<int N>
__device__ __forceinline__ void wait_async()
{
#if defined(__gfx1250__)
    asm volatile("s_wait_asynccnt %0" :: "i"(N) : "memory");
#endif
}

// ---------------------------------------------------------------------------
// Generic bf16 WMMA GEMM:  O = act( A(MxK) @ W(NxK)^T + bias + [add] )
//  - block tile 128x128, BK=32, 256 threads = 8 waves, wave tile 32x64
//  - double-buffered LDS tiles fed by global_load_async_to_lds_b128
//  - A may be f32 in memory (converted + optionally row-masked on the fly;
//    that path stages A via VGPR + ds_store, B still async)
//  - output may be bf16 (feeds next GEMM) or f32 (eb buffer)
// ---------------------------------------------------------------------------
template<int ACT, bool A_F32, bool OUT_F32, bool HAS_ADD, bool HAS_MASK>
__global__ __launch_bounds__(256)
void gemm_bf16_wmma(const void* __restrict__ Av,
                    const bf16* __restrict__ W,
                    const float* __restrict__ bias,
                    const float* __restrict__ add,
                    void* __restrict__ Ov,
                    const unsigned char* __restrict__ mask,
                    int M, int N, int K)
{
    constexpr int BM = 128, BN = 128, BK = 32, LDT = 40; // padded LDS stride
    constexpr int TILE = BM * LDT;
    constexpr int NPREF = A_F32 ? 2 : 4;  // async loads per thread per stage
    __shared__ bf16 As[2 * TILE];
    __shared__ bf16 Bs[2 * TILE];

    const int tid  = threadIdx.x;
    const int lane = tid & 31;
    const int w    = tid >> 5;
    const int wm   = (w >> 1) * 32;   // wave row offset in block tile
    const int wn   = (w & 1) * 64;    // wave col offset in block tile
    const int half = lane >> 4;
    const int l16  = lane & 15;
    const int mBase = blockIdx.y * BM;
    const int nBase = blockIdx.x * BN;

    // staging assignment: this thread owns rows r0 and r0+64, 16B chunk ko0
    const int r0  = tid >> 2;
    const int ko0 = (tid & 3) * 8;

    v8f acc[2][4] = {};

    auto stage = [&](int buf, int kt) {
        bf16* Asb = As + buf * TILE;
        bf16* Bsb = Bs + buf * TILE;
        #pragma unroll
        for (int i = 0; i < 2; i++) {
            const int row = r0 + i * 64;
            // B tile: W is [N,K] row-major == the B^T tile layout we want
            const bf16* Wp = W + (size_t)(nBase + row) * K + kt + ko0;
            async_copy_b128((lds_bf16*)&Bsb[row * LDT + ko0], Wp);

            if (A_F32) {
                const float* Ap = (const float*)Av + (size_t)(mBase + row) * K + kt + ko0;
                float m = 1.0f;
                if (HAS_MASK) m = mask[mBase + row] ? 1.0f : 0.0f;
                float4 f0 = *(const float4*)Ap;
                float4 f1 = *(const float4*)(Ap + 4);
                v8bf v;
                v[0] = (bf16)(f0.x * m); v[1] = (bf16)(f0.y * m);
                v[2] = (bf16)(f0.z * m); v[3] = (bf16)(f0.w * m);
                v[4] = (bf16)(f1.x * m); v[5] = (bf16)(f1.y * m);
                v[6] = (bf16)(f1.z * m); v[7] = (bf16)(f1.w * m);
                *(v8bf*)&Asb[row * LDT + ko0] = v;
            } else {
                const bf16* Ap = (const bf16*)Av + (size_t)(mBase + row) * K + kt + ko0;
                async_copy_b128((lds_bf16*)&Asb[row * LDT + ko0], Ap);
            }
        }
    };

    const int T = K / BK;
    stage(0, 0);
    for (int t = 0; t < T; t++) {
        if (t + 1 < T) {
            stage((t + 1) & 1, (t + 1) * BK);
            wait_async<NPREF>();   // current buffer complete; prefetch in flight
        } else {
            wait_async<0>();
        }
        __syncthreads();

        const bf16* Asb = As + (t & 1) * TILE;
        const bf16* Bsb = Bs + (t & 1) * TILE;

        // ---- build fragments per ISA VGPR layouts ----
        v16bf afr[2], bfr[4];
        #pragma unroll
        for (int i = 0; i < 2; i++) {
            int r = wm + i * 16 + l16;
            // lane half 0: K {0..7, 16..23}; half 1: K {8..15, 24..31}
            v8bf lo = *(const v8bf*)&Asb[r * LDT + half * 8];
            v8bf hi = *(const v8bf*)&Asb[r * LDT + half * 8 + 16];
            afr[i] = __builtin_shufflevector(lo, hi,
                     0,1,2,3,4,5,6,7,8,9,10,11,12,13,14,15);
        }
        #pragma unroll
        for (int j = 0; j < 4; j++) {
            int cc = wn + j * 16 + l16;
            // lane half 0: K {0..7, 16..23}; half 1: K {8..15, 24..31} (column cc)
            v8bf lo = *(const v8bf*)&Bsb[cc * LDT + half * 8];
            v8bf hi = *(const v8bf*)&Bsb[cc * LDT + half * 8 + 16];
            bfr[j] = __builtin_shufflevector(lo, hi,
                     0,1,2,3,4,5,6,7,8,9,10,11,12,13,14,15);
        }

        #pragma unroll
        for (int i = 0; i < 2; i++)
            #pragma unroll
            for (int j = 0; j < 4; j++)
                acc[i][j] = __builtin_amdgcn_wmma_f32_16x16x32_bf16(
                    false, afr[i], false, bfr[j], (short)0, acc[i][j],
                    false, false);
        __syncthreads();
    }

    // ---- epilogue: bias (+addend) + activation, store ----
    #pragma unroll
    for (int i = 0; i < 2; i++) {
        #pragma unroll
        for (int j = 0; j < 4; j++) {
            const int colg = nBase + wn + j * 16 + l16;
            const float bv = bias[colg];
            #pragma unroll
            for (int v = 0; v < 8; v++) {
                const int rowg = mBase + wm + i * 16 + half * 8 + v;
                float val = acc[i][j][v] + bv;
                if (HAS_ADD) val += add[(size_t)rowg * N + colg];
                if (ACT == ACT_RELU) val = val > 0.f ? val : 0.f;
                if (ACT == ACT_TANH) val = tanhf(val);
                if (OUT_F32) ((float*)Ov)[(size_t)rowg * N + colg] = val;
                else         ((bf16*) Ov)[(size_t)rowg * N + colg] = (bf16)val;
            }
        }
    }
}

// fp32 -> bf16 conversion (weights)
__global__ void cvt_bf16_kernel(const float* __restrict__ src,
                                bf16* __restrict__ dst, int n)
{
    int i = blockIdx.x * blockDim.x + threadIdx.x;
    if (i < n) dst[i] = (bf16)src[i];
}

// H0[b,:] = start[:]   (DIM = 512, power of two)
__global__ void init_h_kernel(const float* __restrict__ start,
                              bf16* __restrict__ H, int n)
{
    int i = blockIdx.x * blockDim.x + threadIdx.x;
    if (i < n) H[i] = (bf16)start[i & 511];
}

// HM[b, 0:512] = H[b,:];  HM[b, 512:576] = move_emb[moves[b], :]
__global__ void build_hm_kernel(const bf16* __restrict__ H,
                                const float* __restrict__ move_emb,
                                const int* __restrict__ moves,
                                bf16* __restrict__ HM, int n)
{
    int i = blockIdx.x * blockDim.x + threadIdx.x;
    if (i >= n) return;
    int b = i / 576, c = i % 576;
    HM[i] = (c < 512) ? H[b * 512 + c]
                      : (bf16)move_emb[moves[b] * 64 + (c - 512)];
}

// out[b] = sum_k X[b,k] * w[k] + ob[0]   (one wave per row)
__global__ __launch_bounds__(256)
void rowdot_kernel(const bf16* __restrict__ X, const bf16* __restrict__ w,
                   const float* __restrict__ ob, float* __restrict__ out, int K)
{
    int row  = blockIdx.x * 8 + (threadIdx.x >> 5);
    int lane = threadIdx.x & 31;
    float s = 0.f;
    for (int k = lane; k < K; k += 32)
        s += (float)X[(size_t)row * K + k] * (float)w[k];
    #pragma unroll
    for (int o = 16; o > 0; o >>= 1) s += __shfl_xor(s, o, 32);
    if (lane == 0) out[row] = s + ob[0];
}

// ---------------------------------------------------------------------------
extern "C" void kernel_launch(void* const* d_in, const int* in_sizes, int n_in,
                              void* d_out, int out_size, void* d_ws, size_t ws_size,
                              hipStream_t stream)
{
    (void)in_sizes; (void)n_in; (void)out_size; (void)ws_size;
    constexpr int B = 8192, DIM = 512, EMBED = 768, MED = 64, STEPS = 10;

    const float*         boards   = (const float*)d_in[0];
    const unsigned char* flags    = (const unsigned char*)d_in[1]; // bool array
    const int*           moves    = (const int*)d_in[3];
    const float*         start    = (const float*)d_in[4];
    const float* red_w = (const float*)d_in[5];  const float* red_b = (const float*)d_in[6];
    const float* ca_w  = (const float*)d_in[7];  const float* ca_b  = (const float*)d_in[8];
    const float* cb_w  = (const float*)d_in[9];  const float* cb_b  = (const float*)d_in[10];
    const float* move_emb = (const float*)d_in[11];
    const float* fc1_w = (const float*)d_in[12]; const float* fc1_b = (const float*)d_in[13];
    const float* fc2_w = (const float*)d_in[14]; const float* fc2_b = (const float*)d_in[15];
    const float* fc3_w = (const float*)d_in[16]; const float* fc3_b = (const float*)d_in[17];
    const float* fc4_w = (const float*)d_in[18]; const float* fc4_b = (const float*)d_in[19];
    const float* fc5_w = (const float*)d_in[20]; const float* fc5_b = (const float*)d_in[21];
    const float* out_w = (const float*)d_in[22]; const float* out_b = (const float*)d_in[23];
    float* out = (float*)d_out;

    // workspace carve-up
    char* ws = (char*)d_ws; size_t off = 0;
    auto alloc = [&](size_t bytes) -> char* {
        char* p = ws + off; off = (off + bytes + 255) & ~(size_t)255; return p;
    };
    bf16* redWb = (bf16*)alloc((size_t)DIM * EMBED * 2);
    bf16* caWb  = (bf16*)alloc((size_t)DIM * DIM * 2);
    bf16* cbWb  = (bf16*)alloc((size_t)DIM * DIM * 2);
    bf16* fc1Wb = (bf16*)alloc((size_t)DIM * (DIM + MED) * 2);
    bf16* fc2Wb = (bf16*)alloc((size_t)DIM * DIM * 2);
    bf16* fc3Wb = (bf16*)alloc((size_t)1024 * DIM * 2);
    bf16* fc4Wb = (bf16*)alloc((size_t)1024 * 1024 * 2);
    bf16* fc5Wb = (bf16*)alloc((size_t)DIM * 1024 * 2);
    bf16* outWb = (bf16*)alloc((size_t)DIM * 2);
    bf16* E  = (bf16*)alloc((size_t)B * DIM * 2);
    float* EB = (float*)alloc((size_t)B * DIM * 4);
    bf16* H0 = (bf16*)alloc((size_t)B * DIM * 2);
    bf16* H1 = (bf16*)alloc((size_t)B * DIM * 2);
    bf16* HM = (bf16*)alloc((size_t)B * (DIM + MED) * 2);
    bf16* Xa = (bf16*)alloc((size_t)B * 1024 * 2);
    bf16* Xb = (bf16*)alloc((size_t)B * 1024 * 2);

    auto cvt = [&](const float* s, bf16* d, int n) {
        cvt_bf16_kernel<<<(n + 255) / 256, 256, 0, stream>>>(s, d, n);
    };
    cvt(red_w, redWb, DIM * EMBED);
    cvt(ca_w,  caWb,  DIM * DIM);
    cvt(cb_w,  cbWb,  DIM * DIM);
    cvt(fc1_w, fc1Wb, DIM * (DIM + MED));
    cvt(fc2_w, fc2Wb, DIM * DIM);
    cvt(fc3_w, fc3Wb, 1024 * DIM);
    cvt(fc4_w, fc4Wb, 1024 * 1024);
    cvt(fc5_w, fc5Wb, DIM * 1024);
    cvt(out_w, outWb, DIM);

    init_h_kernel<<<(B * DIM + 255) / 256, 256, 0, stream>>>(start, H0, B * DIM);

    dim3 g512(DIM / 128, B / 128);     // N=512
    dim3 g1024(1024 / 128, B / 128);   // N=1024

    bf16* Hc = H0; bf16* Hn = H1;
    for (int s = 0; s < STEPS; s++) {
        // e_s = relu( mask(boards_s) @ red_w^T + red_b )   [B,512] bf16
        const float* Bp = boards + (size_t)s * B * EMBED;
        if (s < 2)
            gemm_bf16_wmma<ACT_RELU, true, false, false, false>
                <<<g512, 256, 0, stream>>>(Bp, redWb, red_b, nullptr, E,
                                           nullptr, B, DIM, EMBED);
        else
            gemm_bf16_wmma<ACT_RELU, true, false, false, true>
                <<<g512, 256, 0, stream>>>(Bp, redWb, red_b, nullptr, E,
                                           flags + (size_t)(s - 2) * B, B, DIM, EMBED);
        // eb_s = e_s @ cb_w^T + cb_b    [B,512] f32
        gemm_bf16_wmma<ACT_NONE, false, true, false, false>
            <<<g512, 256, 0, stream>>>(E, cbWb, cb_b, nullptr, EB,
                                       nullptr, B, DIM, DIM);
        // h = tanh( h @ ca_w^T + ca_b + eb_s )   [B,512] bf16
        gemm_bf16_wmma<ACT_TANH, false, false, true, false>
            <<<g512, 256, 0, stream>>>(Hc, caWb, ca_b, EB, Hn,
                                       nullptr, B, DIM, DIM);
        bf16* t = Hc; Hc = Hn; Hn = t;
    }

    // concat(h, move_emb[moves])
    build_hm_kernel<<<(B * (DIM + MED) + 255) / 256, 256, 0, stream>>>(
        Hc, move_emb, moves, HM, B * (DIM + MED));

    gemm_bf16_wmma<ACT_RELU, false, false, false, false>
        <<<g512, 256, 0, stream>>>(HM, fc1Wb, fc1_b, nullptr, Xa, nullptr, B, DIM, DIM + MED);
    gemm_bf16_wmma<ACT_RELU, false, false, false, false>
        <<<g512, 256, 0, stream>>>(Xa, fc2Wb, fc2_b, nullptr, Xb, nullptr, B, DIM, DIM);
    gemm_bf16_wmma<ACT_RELU, false, false, false, false>
        <<<g1024, 256, 0, stream>>>(Xb, fc3Wb, fc3_b, nullptr, Xa, nullptr, B, 1024, DIM);
    gemm_bf16_wmma<ACT_RELU, false, false, false, false>
        <<<g1024, 256, 0, stream>>>(Xa, fc4Wb, fc4_b, nullptr, Xb, nullptr, B, 1024, 1024);
    gemm_bf16_wmma<ACT_RELU, false, false, false, false>
        <<<g512, 256, 0, stream>>>(Xb, fc5Wb, fc5_b, nullptr, Xa, nullptr, B, DIM, 1024);

    rowdot_kernel<<<B / 8, 256, 0, stream>>>(Xa, outWb, out_b, out, DIM);
}
